// Classifier_87256555586283
// MI455X (gfx1250) — matrix operand order repaired and checked by reading, not compiled, not run on hardware
//
#include <hip/hip_runtime.h>
#include <hip/hip_bf16.h>
#include <math.h>

// ---------------------------------------------------------------------------
// Problem constants (fixed by the reference harness)
// ---------------------------------------------------------------------------
#define HDIM     768          // hidden width
#define NROWS    16384        // 32 * 512
#define NBASIS   8            // spline bases
#define KEXP     9            // silu + 8 bases per input feature
#define K1       (HDIM * KEXP)        // 6912 = 216 * 32
#define KTILES   (K1 / 32)            // 216
#define NTILES1  (HDIM / 16)          // 48  (16-wide WMMA N tiles)
#define IGROUPS  (HDIM / 32)          // 24  (feature chunks of 32)
#define LDS_PAD  16
#define LDS_STRIDE (KEXP * 32 + LDS_PAD)  // 288 + 16 = 304

typedef __attribute__((ext_vector_type(16))) __bf16 v16bf;
typedef __attribute__((ext_vector_type(8)))  __bf16 v8bf;
typedef __attribute__((ext_vector_type(8)))  float  v8f;

// Workspace layout
#define WF_ELEMS   (KTILES * NTILES1 * 32 * 16)        // 5,308,416 bf16
#define WF_BYTES   (WF_ELEMS * 2)                      // 10,616,832 B (256-aligned)

// ---------------------------------------------------------------------------
// Fused activation expansion: f[0] = silu(x), f[1..8] = cubic B-spline bases.
// Uniform knots t_j = -1 + (j-3)*h, h = 0.4. In knot coordinates u = (x+1)/h+3
// the Cox-de Boor recursion becomes b_p[j] = ((u-j)*b[j] + (j+p+1-u)*b[j+1])/p.
// ---------------------------------------------------------------------------
__device__ __forceinline__ void kan_feats(float x, float* f) {
    f[0] = x / (1.0f + __expf(-x));                    // silu
    const float u = (x + 1.0f) * 2.5f + 3.0f;          // h = 0.4
    float b0[11];
#pragma unroll
    for (int j = 0; j < 11; ++j)
        b0[j] = (u >= (float)j && u < (float)(j + 1)) ? 1.0f : 0.0f;
    float b1[10];
#pragma unroll
    for (int j = 0; j < 10; ++j)
        b1[j] = (u - (float)j) * b0[j] + ((float)(j + 2) - u) * b0[j + 1];
    float b2[9];
#pragma unroll
    for (int j = 0; j < 9; ++j)
        b2[j] = 0.5f * ((u - (float)j) * b1[j] + ((float)(j + 3) - u) * b1[j + 1]);
#pragma unroll
    for (int j = 0; j < 8; ++j)
        f[1 + j] = (1.0f / 3.0f) *
                   ((u - (float)j) * b2[j] + ((float)(j + 4) - u) * b2[j + 1]);
}

// ---------------------------------------------------------------------------
// Kernel 0: build combined layer-1 weights, bf16, pre-swizzled into the exact
// per-lane WMMA B-fragment order so the GEMM streams 32 contiguous bytes/lane
// straight from L2 (10.6 MB total -> fully L2 resident on the 192 MB L2).
// K ordering: k = ig*288 + g*32 + ii  (feature group, basis, feature-in-group)
// Fragment layout: [kt][nt][lane][e]; lane l, elem e -> k_local=(l/16)*16+e,
// n = nt*16 + (l&15).
// ---------------------------------------------------------------------------
__global__ __launch_bounds__(256) void kan_prep_w1(
        const float* __restrict__ base_w1,
        const float* __restrict__ spline_w1,
        const float* __restrict__ scaler1,
        __bf16* __restrict__ Wf) {
    int idx = blockIdx.x * 256 + threadIdx.x;
    if (idx >= WF_ELEMS) return;
    int e    = idx & 15;
    int l    = (idx >> 4) & 31;
    int tile = idx >> 9;            // kt * NTILES1 + nt
    int nt   = tile % NTILES1;
    int kt   = tile / NTILES1;
    int ig   = kt / KEXP;
    int g    = kt % KEXP;
    int kk   = ((l >> 4) << 4) + e; // 0..31 within K-chunk
    int o    = nt * 16 + (l & 15);  // output feature (768)
    int i    = ig * 32 + kk;        // input feature (768)
    float w;
    if (g == 0) {
        w = base_w1[o * HDIM + i];
    } else {
        w = spline_w1[(o * HDIM + i) * NBASIS + (g - 1)] * scaler1[o * HDIM + i];
    }
    Wf[idx] = (__bf16)w;
}

// ---------------------------------------------------------------------------
// Kernel 1: fused KAN layer 1 + erf-GELU.
//   C(16384 x 768) = expand(X) @ W1c^T, bf16 WMMA, f32 accumulate.
//   Block: 256 threads (8 waves), tile 64M x 256N. Wave = 2x4 WMMA tiles
//   (32M x 64N, 64 accumulator VGPRs) -> only 3 N-blocks, so the VALU-heavy
//   spline expansion is duplicated only x3, and each LDS A-fragment feeds
//   4 WMMAs. A expanded on the fly into LDS (64 x 288 bf16). B from L2.
// ---------------------------------------------------------------------------
__global__ __launch_bounds__(256) void kan_layer1(
        const float* __restrict__ X,
        const __bf16* __restrict__ Wf,
        float* __restrict__ H1) {
    __shared__ __bf16 lA[64 * LDS_STRIDE];

    const int tid  = threadIdx.x;
    const int lane = tid & 31;
    const int wid  = tid >> 5;
    const int wm   = wid >> 2;             // 0..1  (M direction, 32 rows each)
    const int wn   = wid & 3;              // 0..3  (N direction, 64 cols each)
    const int M0   = blockIdx.y * 64;      // row base
    const int NT0  = blockIdx.x * 16;      // 16-wide N-tile base (16 tiles=256)

    const int xi = tid & 31;               // feature within 32-chunk
    const int xr = tid >> 5;               // row group base (8 rows apart)

    // A-fragment addressing (contiguous k order used by this code, same as Wf packing, wave32):
    // lanes 0-15: row M=lane, K = {0..15}; lanes 16-31: K = {16..31}
    const int rowA  = lane & 15;
    const int koff0 = (lane >> 4) ? 16 : 0;
    const int koff1 = koff0 + 8;

    v8f acc[2][4] = {};

    for (int ig = 0; ig < IGROUPS; ++ig) {
        __syncthreads();   // protect lA against previous iteration's readers
        // ---- expand 64 rows x 32 features -> 9 bf16 features each, into LDS
#pragma unroll
        for (int q = 0; q < 8; ++q) {
            int r = xr + q * 8;                         // 0..63
            float x = X[(M0 + r) * HDIM + ig * 32 + xi];
            if (ig + 1 < IGROUPS)                       // global_prefetch_b8
                __builtin_prefetch(&X[(M0 + r) * HDIM + (ig + 1) * 32 + xi], 0, 3);
            float f[KEXP];
            kan_feats(x, f);
#pragma unroll
            for (int g = 0; g < KEXP; ++g)
                lA[r * LDS_STRIDE + g * 32 + xi] = (__bf16)f[g];
        }
        __syncthreads();
        // ---- 9 WMMA K-steps over this chunk
#pragma unroll
        for (int g = 0; g < KEXP; ++g) {
            const int kt = ig * KEXP + g;
            v16bf bfrag[4];
#pragma unroll
            for (int b = 0; b < 4; ++b) {
                int nt = NT0 + wn * 4 + b;
                bfrag[b] = ((const v16bf*)Wf)[(kt * NTILES1 + nt) * 32 + lane];
            }
#pragma unroll
            for (int a = 0; a < 2; ++a) {
                int r = wm * 32 + a * 16 + rowA;
                union { v16bf v; v8bf h[2]; } au;
                au.h[0] = *(const v8bf*)&lA[r * LDS_STRIDE + g * 32 + koff0];
                au.h[1] = *(const v8bf*)&lA[r * LDS_STRIDE + g * 32 + koff1];
#pragma unroll
                for (int b = 0; b < 4; ++b)
                    acc[a][b] = __builtin_amdgcn_wmma_f32_16x16x32_bf16(
                        false, au.v, false, bfrag[b], (short)0, acc[a][b],
                        false, false);
            }
        }
    }

    // ---- epilogue: exact (erf) GELU, store f32 intermediate
    // C layout: VGPR r, lanes 0-15 -> M=r, N=lane; lanes 16-31 -> M=r+8, N=lane-16
#pragma unroll
    for (int a = 0; a < 2; ++a)
#pragma unroll
    for (int b = 0; b < 4; ++b)
#pragma unroll
    for (int r = 0; r < 8; ++r) {
        int row = M0 + wm * 32 + a * 16 + r + ((lane >> 4) << 3);
        int col = blockIdx.x * 256 + wn * 64 + b * 16 + (lane & 15);
        float v = acc[a][b][r];
        float gl = 0.5f * v * (1.0f + erff(v * 0.70710678118654752f));
        H1[row * HDIM + col] = gl;
    }
}

// ---------------------------------------------------------------------------
// Kernel 2: KAN layer 2 (out dim = 2). Memory/VALU bound -> fused dot product.
// Block = 256 threads = 4 rows x 64 threads; LDS reduction per row.
// ---------------------------------------------------------------------------
__global__ __launch_bounds__(256) void kan_layer2(
        const float* __restrict__ H1,
        const float* __restrict__ base_w2,
        const float* __restrict__ spline_w2,
        const float* __restrict__ scaler2,
        float* __restrict__ out) {
    __shared__ float red[256][2];
    const int rg  = threadIdx.x >> 6;          // 0..3 row group
    const int lg  = threadIdx.x & 63;
    const int row = blockIdx.x * 4 + rg;

    float a0 = 0.0f, a1 = 0.0f;
#pragma unroll
    for (int it = 0; it < HDIM / 64; ++it) {
        int i = lg + it * 64;
        float x = H1[row * HDIM + i];
        float f[KEXP];
        kan_feats(x, f);
        float s0 = 0.0f, s1 = 0.0f;
#pragma unroll
        for (int g = 0; g < NBASIS; ++g) {
            s0 += f[1 + g] * spline_w2[(0 * HDIM + i) * NBASIS + g];
            s1 += f[1 + g] * spline_w2[(1 * HDIM + i) * NBASIS + g];
        }
        a0 += f[0] * base_w2[i]        + scaler2[i]        * s0;
        a1 += f[0] * base_w2[HDIM + i] + scaler2[HDIM + i] * s1;
    }
    red[threadIdx.x][0] = a0;
    red[threadIdx.x][1] = a1;
    __syncthreads();
    if (lg < 2) {
        float s = 0.0f;
        for (int t = 0; t < 64; ++t) s += red[rg * 64 + t][lg];
        out[row * 2 + lg] = s;
    }
}

// ---------------------------------------------------------------------------
// Host launcher
// ---------------------------------------------------------------------------
extern "C" void kernel_launch(void* const* d_in, const int* in_sizes, int n_in,
                              void* d_out, int out_size, void* d_ws, size_t ws_size,
                              hipStream_t stream) {
    const float* hidden    = (const float*)d_in[0];
    const float* base_w1   = (const float*)d_in[1];
    const float* spline_w1 = (const float*)d_in[2];
    const float* scaler1   = (const float*)d_in[3];
    const float* base_w2   = (const float*)d_in[4];
    const float* spline_w2 = (const float*)d_in[5];
    const float* scaler2   = (const float*)d_in[6];
    float* out = (float*)d_out;

    char*   ws = (char*)d_ws;
    __bf16* Wf = (__bf16*)ws;                       // 10.6 MB swizzled weights
    float*  H1 = (float*)(ws + WF_BYTES);           // 48 MB f32 intermediate

    // 1) combine + swizzle layer-1 weights into WMMA fragment order
    kan_prep_w1<<<(WF_ELEMS + 255) / 256, 256, 0, stream>>>(
        base_w1, spline_w1, scaler1, Wf);

    // 2) fused expand -> bf16 WMMA GEMM -> GELU
    dim3 g1(HDIM / 256, NROWS / 64);                // (3, 256)
    kan_layer1<<<g1, 256, 0, stream>>>(hidden, Wf, H1);

    // 3) tiny output layer (out dim 2), fused VALU dot products
    kan_layer2<<<NROWS / 4, 256, 0, stream>>>(H1, base_w2, spline_w2, scaler2, out);
}
